// AMRLMax_26757646254325
// MI455X (gfx1250) — compile-verified, matrix operand order checked by reading, not run on hardware
//
#include <hip/hip_runtime.h>
#include <hip/hip_bf16.h>
#include <math.h>

#define B_   128
#define T_   512
#define I_   256
#define H_   512
#define G4H  2048            // 4*H
#define KK   768             // I + H
#define NBLK 64              // persistent grid
#define SP   776             // padded LDS row stride (elements) -> conflict-minimal
#define SMEM_BYTES (64 * SP * sizeof(__bf16))   // 64 weight rows resident in LDS

typedef __attribute__((ext_vector_type(8)))  float  v8f;
typedef __attribute__((ext_vector_type(8)))  __bf16 v8bf;
typedef __attribute__((ext_vector_type(16))) __bf16 v16bf;

union BFPack { v16bf v; v8bf h[2]; };

// ---------------- prep kernels ----------------

// Fuse W_ih [2048 x 256] and W_hh [2048 x 512] into bf16 Wc [2048 x 768] row-major.
__global__ void lstm_prep_weights(const float* __restrict__ Wih,
                                  const float* __restrict__ Whh,
                                  __bf16* __restrict__ Wc) {
    int idx = blockIdx.x * blockDim.x + threadIdx.x;
    if (idx >= G4H * KK) return;
    int n = idx / KK;
    int k = idx - n * KK;
    float v = (k < I_) ? Wih[n * I_ + k] : Whh[n * H_ + (k - I_)];
    Wc[idx] = (__bf16)v;
}

// x [B][T][I] f32 -> xT [T][B][I] bf16
__global__ void lstm_prep_x(const float* __restrict__ x, __bf16* __restrict__ xT) {
    size_t idx = (size_t)blockIdx.x * blockDim.x + threadIdx.x;
    if (idx >= (size_t)T_ * B_ * I_) return;
    int i  = (int)(idx % I_);
    size_t tb = idx / I_;
    int b  = (int)(tb % B_);
    int t  = (int)(tb / B_);
    xT[idx] = (__bf16)x[((size_t)b * T_ + t) * I_ + i];
}

// h0 -> bf16 h buffer A; reset grid-barrier state
__global__ void lstm_prep_state(const float* __restrict__ h0,
                                __bf16* __restrict__ hA,
                                unsigned* __restrict__ bar) {
    int idx = blockIdx.x * blockDim.x + threadIdx.x;
    if (idx < B_ * H_) hA[idx] = (__bf16)h0[idx];
    if (idx == 0) { bar[0] = 0u; bar[1] = 0u; }
}

// ---------------- device helpers ----------------

__device__ __forceinline__ v16bf ld_pack(const __bf16* p) {
    BFPack a;
    a.h[0] = *(const v8bf*)(p);
    a.h[1] = *(const v8bf*)(p + 16);
    return a.v;
}

__device__ __forceinline__ void grid_barrier(unsigned* __restrict__ bar) {
    __syncthreads();
    if (threadIdx.x == 0) {
        __builtin_amdgcn_fence(__ATOMIC_RELEASE, "agent");
        unsigned gen = __hip_atomic_load(&bar[1], __ATOMIC_RELAXED, __HIP_MEMORY_SCOPE_AGENT);
        unsigned arrived = __hip_atomic_fetch_add(&bar[0], 1u, __ATOMIC_ACQ_REL,
                                                  __HIP_MEMORY_SCOPE_AGENT) + 1u;
        if (arrived == (unsigned)NBLK) {
            __hip_atomic_store(&bar[0], 0u, __ATOMIC_RELAXED, __HIP_MEMORY_SCOPE_AGENT);
            __hip_atomic_store(&bar[1], gen + 1u, __ATOMIC_RELEASE, __HIP_MEMORY_SCOPE_AGENT);
        } else {
            while (__hip_atomic_load(&bar[1], __ATOMIC_ACQUIRE, __HIP_MEMORY_SCOPE_AGENT) == gen)
                __builtin_amdgcn_s_sleep(2);
        }
        __builtin_amdgcn_fence(__ATOMIC_ACQUIRE, "agent");
    }
    __syncthreads();
}

// ---------------- persistent LSTM kernel ----------------

__global__ __launch_bounds__(128)
void lstm_persistent(const __bf16* __restrict__ xT,   // [T][B][I] bf16
                     const __bf16* __restrict__ Wc,   // [2048][768] bf16
                     const float*  __restrict__ bias, // [2048]
                     const float*  __restrict__ c0,   // [B][H]
                     __bf16*       __restrict__ hA,   // [B][H] bf16 (holds h0 at entry)
                     __bf16*       __restrict__ hB,   // [B][H] bf16
                     float*        __restrict__ out,  // concat_out | hT | cT
                     unsigned*     __restrict__ bar) {
    extern __shared__ __bf16 smem[];                  // 64 rows x SP

    const int lane  = threadIdx.x & 31;
    const int wave  = threadIdx.x >> 5;
    const int jtile = blockIdx.x >> 1;                // 32 jtiles; shared by all 4 waves
    const int mhalf = blockIdx.x & 1;
    const int m0    = (mhalf * 4 + wave) * 16;        // batch tile
    const int j0    = jtile * 16;                     // hidden tile

    const int lrow = lane & 15;
    const int dk   = (lane >> 4) * 8;                 // K sub-offset per ISA A/B layout

    // ---- stage this block's 64 weight rows (4 gates x 16) into LDS, once ----
    for (int c = threadIdx.x; c < 64 * 96; c += 128) {
        int row = c / 96, c16 = c - row * 96;
        int gate = row >> 4, rr = row & 15;
        const __bf16* src = Wc + (size_t)(gate * H_ + j0 + rr) * KK + c16 * 8;
        *(v8bf*)(smem + row * SP + c16 * 8) = *(const v8bf*)src;
    }
    __syncthreads();

    const __bf16* bs0 = smem + (0 * 16 + lrow) * SP + dk;
    const __bf16* bs1 = smem + (1 * 16 + lrow) * SP + dk;
    const __bf16* bs2 = smem + (2 * 16 + lrow) * SP + dk;
    const __bf16* bs3 = smem + (3 * 16 + lrow) * SP + dk;

    // ---- per-lane persistent state: cell, running max, bias ----
    const int j     = j0 + lrow;                      // hidden channel owned by lane
    const int mbase = m0 + ((lane >> 4) << 3);        // C/D layout: lanes 16-31 hold M+8
    const bool firstHalf = (j0 < H_ / 2);             // uniform per wave

    float cc[8], rm[8];
    #pragma unroll
    for (int r = 0; r < 8; ++r) {
        cc[r] = c0[(mbase + r) * H_ + j];
        rm[r] = -INFINITY;
    }
    const float bi = bias[0 * H_ + j];
    const float bf = bias[1 * H_ + j];
    const float bg = bias[2 * H_ + j];
    const float bo = bias[3 * H_ + j];

    // ---- time recurrence ----
    #pragma unroll 1
    for (int t = 0; t < T_; ++t) {
        const __bf16* hp = (t & 1) ? hB : hA;         // read buffer
        __bf16*       hn = (t & 1) ? hA : hB;         // write buffer

        const __bf16* xrow = xT + ((size_t)t * B_ + (m0 + lrow)) * I_ + dk;
        const __bf16* hrow = hp + (size_t)(m0 + lrow) * H_ + dk;

        v8f ai = {}, af = {}, ag = {}, ao = {};

        v16bf acur = ld_pack(xrow);                   // slab 0 prefetched
        #pragma unroll
        for (int s = 0; s < 24; ++s) {
            const int kk = s * 32;
            v16bf anxt = acur;
            if (s + 1 < 24) {                         // prefetch next A slab (global)
                const __bf16* ap = (s + 1 < 8) ? (xrow + (s + 1) * 32)
                                               : (hrow + (s + 1 - 8) * 32);
                anxt = ld_pack(ap);
            }
            // B operands from LDS (ds_load_b128), 4 gate chains for ILP
            ai = __builtin_amdgcn_wmma_f32_16x16x32_bf16(false, acur, false, ld_pack(bs0 + kk), (short)0, ai, false, false);
            af = __builtin_amdgcn_wmma_f32_16x16x32_bf16(false, acur, false, ld_pack(bs1 + kk), (short)0, af, false, false);
            ag = __builtin_amdgcn_wmma_f32_16x16x32_bf16(false, acur, false, ld_pack(bs2 + kk), (short)0, ag, false, false);
            ao = __builtin_amdgcn_wmma_f32_16x16x32_bf16(false, acur, false, ld_pack(bs3 + kk), (short)0, ao, false, false);
            acur = anxt;
        }

        // ---- fused epilogue: gates -> c,h, cummax, stores ----
        #pragma unroll
        for (int r = 0; r < 8; ++r) {
            const int m = mbase + r;
            const float si = 1.0f / (1.0f + __expf(-(ai[r] + bi)));
            const float sf = 1.0f / (1.0f + __expf(-(af[r] + bf)));
            const float tg = tanhf(ag[r] + bg);
            const float so = 1.0f / (1.0f + __expf(-(ao[r] + bo)));

            const float cnew = sf * cc[r] + si * tg;
            cc[r] = cnew;
            const float h = so * tanhf(cnew);

            hn[m * H_ + j] = (__bf16)h;

            float* outp = out + ((size_t)m * T_ + t) * H_;
            if (firstHalf) {
                rm[r] = fmaxf(rm[r], h);
                __builtin_nontemporal_store(rm[r], &outp[j]);   // cummax half
            } else {
                __builtin_nontemporal_store(h, &outp[j]);       // pass-through half
            }
            if (t == T_ - 1) {
                out[(size_t)B_ * T_ * H_ + (size_t)m * H_ + j] = h;                      // hT
                out[(size_t)B_ * T_ * H_ + (size_t)B_ * H_ + (size_t)m * H_ + j] = cnew; // cT
            }
        }

        grid_barrier(bar);                            // step t visible everywhere
    }
}

// ---------------- host launch ----------------

extern "C" void kernel_launch(void* const* d_in, const int* in_sizes, int n_in,
                              void* d_out, int out_size, void* d_ws, size_t ws_size,
                              hipStream_t stream) {
    const float* x    = (const float*)d_in[0];  // [B,T,I]
    const float* h0   = (const float*)d_in[1];  // [B,H]
    const float* c0   = (const float*)d_in[2];  // [B,H]
    const float* Wih  = (const float*)d_in[3];  // [4H,I]
    const float* Whh  = (const float*)d_in[4];  // [4H,H]
    const float* bias = (const float*)d_in[5];  // [4H]
    float* out = (float*)d_out;

    char* ws = (char*)d_ws;
    size_t off = 0;
    auto carve = [&](size_t bytes) -> void* {
        void* p = ws + off;
        off = (off + bytes + 255) & ~(size_t)255;
        return p;
    };
    __bf16*   Wc  = (__bf16*)carve((size_t)G4H * KK * sizeof(__bf16));     // 3 MB
    __bf16*   xT  = (__bf16*)carve((size_t)T_ * B_ * I_ * sizeof(__bf16)); // 33.5 MB
    __bf16*   hA  = (__bf16*)carve((size_t)B_ * H_ * sizeof(__bf16));
    __bf16*   hB  = (__bf16*)carve((size_t)B_ * H_ * sizeof(__bf16));
    unsigned* bar = (unsigned*)carve(2 * sizeof(unsigned));

    {
        int n = G4H * KK;
        lstm_prep_weights<<<(n + 255) / 256, 256, 0, stream>>>(Wih, Whh, Wc);
    }
    {
        size_t n = (size_t)T_ * B_ * I_;
        lstm_prep_x<<<(unsigned)((n + 255) / 256), 256, 0, stream>>>(x, xT);
    }
    {
        int n = B_ * H_;
        lstm_prep_state<<<(n + 255) / 256, 256, 0, stream>>>(h0, hA, bar);
    }

    // one persistent launch runs all 512 steps with device-wide barriers
    lstm_persistent<<<NBLK, 128, SMEM_BYTES, stream>>>(xT, Wc, bias, c0, hA, hB, out, bar);
}
